// OutputBlock_3736621547805
// MI455X (gfx1250) — compile-verified, matrix operand order checked by reading, not run on hardware
//
#include <hip/hip_runtime.h>
#include <hip/hip_bf16.h>

typedef __attribute__((ext_vector_type(16))) _Float16 v16h;
typedef __attribute__((ext_vector_type(8)))  _Float16 v8h;
typedef __attribute__((ext_vector_type(4)))  _Float16 v4h;
typedef __attribute__((ext_vector_type(8)))  float    v8f;

#define HID   512
#define HHALF 256
#define NROWS 200000
#define NGR   2000
#define ROWS_PER_BLOCK 128
#define LDS_STRIDE 520   // 512 + 8 f16 pad -> 1040B rows, 16B aligned, bank-conflict-free

// ---- prep: fp32 -> f16 copies of W1 and Ww into workspace; zero the output ----
__global__ void prep_kernel(const float* __restrict__ W1, const float* __restrict__ Ww,
                            float* __restrict__ out, _Float16* __restrict__ W1h,
                            _Float16* __restrict__ Wwh) {
    int i = blockIdx.x * 256 + threadIdx.x;
    if (i < HHALF * HID) W1h[i] = (_Float16)W1[i];
    if (i < HID)         Wwh[i] = (_Float16)Ww[i];
    if (i < NGR)         out[i] = 0.0f;
}

// ---- fused: alpha = h@Ww+bw ; y = silu(h@W1^T+b1)@W2+b2 ; out[batch] += y*alpha ----
__global__ void __launch_bounds__(256)
fused_energy_kernel(const float* __restrict__ h,
                    const long long* __restrict__ batch,
                    const _Float16* __restrict__ W1h,
                    const _Float16* __restrict__ Wwh,
                    const float* __restrict__ b1,
                    const float* __restrict__ W2,
                    const float* __restrict__ b2,
                    const float* __restrict__ bw,
                    float* __restrict__ out) {
    extern __shared__ _Float16 hLds[];   // ROWS_PER_BLOCK * LDS_STRIDE f16

    const int tid  = threadIdx.x;
    const int row0 = blockIdx.x * ROWS_PER_BLOCK;

    // ---- stage h tile: fp32 global -> f16 LDS (coalesced float4 loads) ----
    #pragma unroll 4
    for (int i = 0; i < (ROWS_PER_BLOCK * HID / 4) / 256; ++i) {
        int idx  = i * 256 + tid;           // 16384 float4 chunks
        int r    = idx >> 7;                // /128: local row
        int c4   = (idx & 127) << 2;        // column (multiple of 4)
        int grow = row0 + r;
        float4 v = make_float4(0.f, 0.f, 0.f, 0.f);
        if (grow < NROWS)
            v = *(const float4*)(h + (size_t)grow * HID + c4);
        v4h p = { (_Float16)v.x, (_Float16)v.y, (_Float16)v.z, (_Float16)v.w };
        *(v4h*)(&hLds[r * LDS_STRIDE + c4]) = p;
    }
    __syncthreads();

    // ---- per-wave WMMA: wave w owns local rows [w*16, w*16+16) ----
    const int lane  = tid & 31;
    const int wav   = tid >> 5;
    const int lm    = lane & 15;            // M (A) / N (B) sub-index
    const int khalf = (lane >> 4) << 3;     // lanes 0-15: K base 0 ; lanes 16-31: K base 8
    const _Float16* aRow = &hLds[(wav * 16 + lm) * LDS_STRIDE];

    // alpha pass: B columns all replicate Ww -> acc[r] = alpha of row m (C layout)
    v8f aacc = {};
    #pragma unroll
    for (int k = 0; k < HID / 32; ++k) {
        int k0 = k * 32 + khalf;
        v8h alo = *(const v8h*)(aRow + k0);
        v8h ahi = *(const v8h*)(aRow + k0 + 16);
        v16h a  = __builtin_shufflevector(alo, ahi, 0,1,2,3,4,5,6,7,8,9,10,11,12,13,14,15);
        v8h blo = *(const v8h*)(Wwh + k0);
        v8h bhi = *(const v8h*)(Wwh + k0 + 16);
        v16h b  = __builtin_shufflevector(blo, bhi, 0,1,2,3,4,5,6,7,8,9,10,11,12,13,14,15);
        aacc = __builtin_amdgcn_wmma_f32_16x16x32_f16(false, a, false, b,
                                                      (short)0, aacc, false, false);
    }

    // main pass: 16 N-tiles of W1^T, fused silu + W2 reduction
    float yp[8] = {0.f, 0.f, 0.f, 0.f, 0.f, 0.f, 0.f, 0.f};
    for (int nt = 0; nt < HHALF / 16; ++nt) {
        const _Float16* bRow = W1h + (nt * 16 + lm) * HID;   // W1[n][k] == B[k][n]
        v8f c = {};
        #pragma unroll
        for (int k = 0; k < HID / 32; ++k) {
            int k0 = k * 32 + khalf;
            v8h alo = *(const v8h*)(aRow + k0);
            v8h ahi = *(const v8h*)(aRow + k0 + 16);
            v16h a  = __builtin_shufflevector(alo, ahi, 0,1,2,3,4,5,6,7,8,9,10,11,12,13,14,15);
            v8h blo = *(const v8h*)(bRow + k0);
            v8h bhi = *(const v8h*)(bRow + k0 + 16);
            v16h b  = __builtin_shufflevector(blo, bhi, 0,1,2,3,4,5,6,7,8,9,10,11,12,13,14,15);
            c = __builtin_amdgcn_wmma_f32_16x16x32_f16(false, a, false, b,
                                                       (short)0, c, false, false);
        }
        int   ncol = nt * 16 + lm;
        float w2v  = W2[ncol];
        float b1v  = b1[ncol];
        #pragma unroll
        for (int r = 0; r < 8; ++r) {
            float t = c[r] + b1v;
            float s = t * (1.0f / (1.0f + __expf(-t)));      // SiLU
            yp[r] += s * w2v;
        }
    }

    // reduce W2 dot across the 16 lanes sharing each M row (stays in wave32 halves)
    #pragma unroll
    for (int r = 0; r < 8; ++r) {
        float v = yp[r];
        v += __shfl_xor(v, 1, 32);
        v += __shfl_xor(v, 2, 32);
        v += __shfl_xor(v, 4, 32);
        v += __shfl_xor(v, 8, 32);
        yp[r] = v;
    }

    // gate by alpha and scatter into segment sums
    const float b2v = b2[0];
    const float bwv = bw[0];
    if (lm == 0) {                                  // lanes 0 (M=r) and 16 (M=r+8)
        int mbase = row0 + wav * 16 + khalf;        // khalf==8 exactly for lane 16
        #pragma unroll
        for (int r = 0; r < 8; ++r) {
            int row = mbase + r;
            if (row < NROWS) {
                int g = (int)batch[row];
                atomicAdd(out + g, (yp[r] + b2v) * (aacc[r] + bwv));
            }
        }
    }
}

extern "C" void kernel_launch(void* const* d_in, const int* in_sizes, int n_in,
                              void* d_out, int out_size, void* d_ws, size_t ws_size,
                              hipStream_t stream) {
    // setup_inputs order:
    // 0:h 1:edge_index 2:edge_weight 3:batch 4:alpha 5:W1 6:b1 7:W2 8:b2 9:Ww 10:bw
    const float*     h     = (const float*)d_in[0];
    const long long* batch = (const long long*)d_in[3];
    const float*     W1    = (const float*)d_in[5];
    const float*     b1    = (const float*)d_in[6];
    const float*     W2    = (const float*)d_in[7];
    const float*     b2    = (const float*)d_in[8];
    const float*     Ww    = (const float*)d_in[9];
    const float*     bw    = (const float*)d_in[10];
    float*           out   = (float*)d_out;
    (void)in_sizes; (void)n_in; (void)out_size; (void)ws_size;

    _Float16* W1h = (_Float16*)d_ws;                  // 256*512 f16
    _Float16* Wwh = W1h + HHALF * HID;                // 512 f16

    prep_kernel<<<(HHALF * HID + 255) / 256, 256, 0, stream>>>(W1, Ww, out, W1h, Wwh);

    const int    grid = (NROWS + ROWS_PER_BLOCK - 1) / ROWS_PER_BLOCK;
    const size_t lds  = (size_t)ROWS_PER_BLOCK * LDS_STRIDE * sizeof(_Float16); // 130 KB
    fused_energy_kernel<<<grid, 256, lds, stream>>>(h, batch, W1h, Wwh, b1, W2, b2, bw, out);
}
